// GemmaAttention_45320494907672
// MI455X (gfx1250) — matrix operand order repaired and checked by reading, not compiled, not run on hardware
//
#include <hip/hip_runtime.h>
#include <hip/hip_bf16.h>

#define B_ 2
#define S_ 2048
#define D_ 2048
#define H_ 8
#define HD_ 256
#define WIN_ 1024
#define SCALE_ 0.0625f  /* 1/sqrt(256) */

typedef unsigned short ushort_t;
typedef unsigned int uint_t;

typedef __attribute__((ext_vector_type(16))) __bf16 v16bf;
typedef __attribute__((ext_vector_type(8)))  float  v8f;

union FragU { uint4 q[2]; v16bf v; };

__device__ inline ushort_t f2bf(float f) {
  uint_t u = __float_as_uint(f);
  uint_t r = u + 0x7FFFu + ((u >> 16) & 1u);
  return (ushort_t)(r >> 16);
}
__device__ inline float bf2f(ushort_t h) {
  return __uint_as_float(((uint_t)h) << 16);
}

// CDNA5 async copy: LDS[lds_off .. +15] = MEM[gaddr .. +15], per lane (ASYNCcnt).
__device__ inline void async_load_b128(uint_t lds_off, const ushort_t* gaddr) {
  asm volatile("global_load_async_to_lds_b128 %0, %1, off"
               :: "v"(lds_off), "v"((const void*)gaddr)
               : "memory");
}
__device__ inline void wait_async0() {
  asm volatile("s_wait_asynccnt 0" ::: "memory");
}
__device__ inline uint_t lds_off32(const void* p) {
  return (uint_t)(uintptr_t)p;  // generic LDS ptr: addr[31:0] == in-allocation offset
}

// A-fragment (16x32 MxK): lane row = L%16, hi = L>=16; elem e -> K = (e<8?e:e+8)+8*hi
__device__ inline v16bf load_a_frag(const ushort_t* rowbase, int hi) {
  FragU f;
  f.q[0] = *(const uint4*)(rowbase + 8 * hi);
  f.q[1] = *(const uint4*)(rowbase + 16 + 8 * hi);
  return f.v;
}
// B-fragment (32x16 KxN): lane col = L%16, hi = L>=16; elem e -> K = (e<8?e:e+8)+8*hi
__device__ inline v16bf load_b_frag(const ushort_t* colbase, int hi) {
  FragU f;
  f.q[0] = *(const uint4*)(colbase + 8 * hi);
  f.q[1] = *(const uint4*)(colbase + 16 + 8 * hi);
  return f.v;
}

// Bulk f32 -> bf16 (RNE), 4 elems/thread. n must be a multiple of 4.
__global__ __launch_bounds__(256) void cvt_bf16_kernel(const float* __restrict__ src,
                                                       ushort_t* __restrict__ dst, int n) {
  const int i = (blockIdx.x * blockDim.x + threadIdx.x) * 4;
  if (i >= n) return;
  float4 f = *(const float4*)(src + i);
  ushort_t t[4] = {f2bf(f.x), f2bf(f.y), f2bf(f.z), f2bf(f.w)};
  *(uint2*)(dst + i) = *(const uint2*)t;
}

// C = A(MxK,bf16) * W(NxK,bf16)^T.  Block tile 256x128x32, 8 waves (each 32Mx128N).
// Double-buffered LDS; next k-tile streamed with global_load_async_to_lds_b128
// while the current tile feeds v_wmma.
template <bool CF32>
__global__ __launch_bounds__(256) void gemm_wmma_kernel(
    const ushort_t* __restrict__ A, const ushort_t* __restrict__ W,
    void* __restrict__ Cptr, int M, int N, int K) {
  __shared__ ushort_t lA[2][256 * 40];
  __shared__ ushort_t lB[2][128 * 40];
  const int m0 = blockIdx.x * 256;
  const int n0 = blockIdx.y * 128;
  const int tid = threadIdx.x;
  const int lane = tid & 31;
  const int l16 = lane & 15;
  const int hi = lane >> 4;
  const int wv = tid >> 5;  // 0..7, each wave owns 32 M-rows x 128 N-cols

  v8f acc[2][8];
#pragma unroll
  for (int i = 0; i < 2; ++i)
#pragma unroll
    for (int j = 0; j < 8; ++j)
#pragma unroll
      for (int e = 0; e < 8; ++e) acc[i][j][e] = 0.0f;

  const int ra = tid;             // A row within tile (0..255)
  const int rb = tid >> 1;        // B row within tile (0..127)
  const int cs = (tid & 1) << 4;  // 0 or 16
  const int nk = K >> 5;

  const ushort_t* Abase = A + (size_t)(m0 + ra) * K;
  const ushort_t* Wbase = W + (size_t)(n0 + rb) * K + cs;

  auto stage = [&](int buf, int k0) {
    const ushort_t* sa = Abase + k0;
    async_load_b128(lds_off32(&lA[buf][ra * 40 + 0]), sa + 0);
    async_load_b128(lds_off32(&lA[buf][ra * 40 + 8]), sa + 8);
    async_load_b128(lds_off32(&lA[buf][ra * 40 + 16]), sa + 16);
    async_load_b128(lds_off32(&lA[buf][ra * 40 + 24]), sa + 24);
    const ushort_t* sb = Wbase + k0;
    async_load_b128(lds_off32(&lB[buf][rb * 40 + cs]), sb + 0);
    async_load_b128(lds_off32(&lB[buf][rb * 40 + cs + 8]), sb + 8);
  };

  stage(0, 0);
  wait_async0();
  __syncthreads();

  for (int kt = 0; kt < nk; ++kt) {
    const int buf = kt & 1;
    if (kt + 1 < nk) stage(buf ^ 1, (kt + 1) << 5);  // overlap with WMMA below

    v16bf af[2], bfv[8];
#pragma unroll
    for (int i = 0; i < 2; ++i)
      af[i] = load_a_frag(&lA[buf][(wv * 32 + i * 16 + l16) * 40], hi);
#pragma unroll
    for (int j = 0; j < 8; ++j)
      bfv[j] = load_b_frag(&lB[buf][(j * 16 + l16) * 40], hi);
#pragma unroll
    for (int i = 0; i < 2; ++i)
#pragma unroll
      for (int j = 0; j < 8; ++j)
        acc[i][j] = __builtin_amdgcn_wmma_f32_16x16x32_bf16(
            false, af[i], false, bfv[j], (short)0, acc[i][j], false, false);

    wait_async0();
    __syncthreads();
  }

#pragma unroll
  for (int i = 0; i < 2; ++i) {
#pragma unroll
    for (int j = 0; j < 8; ++j) {
      const int col = n0 + j * 16 + l16;
#pragma unroll
      for (int rr = 0; rr < 8; ++rr) {
        const int row = m0 + wv * 32 + i * 16 + rr + 8 * hi;
        const float v = acc[i][j][rr];
        if (CF32) ((float*)Cptr)[(size_t)row * N + col] = v;
        else ((ushort_t*)Cptr)[(size_t)row * N + col] = f2bf(v);
      }
    }
  }
}

// In-place RoPE on bf16 [B*S][nheads*256]; pair (j, j+128) per thread.
__global__ __launch_bounds__(256) void rope_kernel(ushort_t* X, const int* pos_ids,
                                                   int nheads, int total) {
  int idx = blockIdx.x * blockDim.x + threadIdx.x;
  if (idx >= total) return;
  const int j = idx & 127;
  const int t = idx >> 7;
  const int h = t % nheads;
  const int row = t / nheads;  // b*S + s
  const float pos = (float)pos_ids[row];
  const float f = pos * __expf(-(float)j * (9.21034037197618f / 128.0f));  // 10000^(-j/128)
  float sn, cs;
  __sincosf(f, &sn, &cs);
  const size_t base = (size_t)row * (nheads * 256) + (size_t)h * 256;
  const float x1 = bf2f(X[base + j]);
  const float x2 = bf2f(X[base + j + 128]);
  X[base + j] = f2bf(x1 * cs - x2 * sn);
  X[base + j + 128] = f2bf(x2 * cs + x1 * sn);
}

// Sliding-window causal flash attention. 1 block = (b, h, 64 query rows), 4 waves.
__global__ __launch_bounds__(128) void attn_kernel(
    const ushort_t* __restrict__ Qb, const ushort_t* __restrict__ Kb,
    const ushort_t* __restrict__ Vb, ushort_t* __restrict__ AO) {
  __shared__ ushort_t lK[32 * 264];   // [key][hd], padded
  __shared__ ushort_t lVt[256 * 40];  // [hd][key], padded (transposed V)
  __shared__ float lS[4][16][33];
  __shared__ ushort_t lP[4][16 * 40];
  __shared__ float lAlpha[4][16];
  __shared__ float lL[4][16];

  const int blk = blockIdx.x;
  const int qb = blk & 31;
  const int h = (blk >> 5) & 7;
  const int b = blk >> 8;
  const int q0 = qb * 64;

  const int tid = threadIdx.x;
  const int lane = tid & 31;
  const int l16 = lane & 15;
  const int hi = lane >> 4;
  const int wv = tid >> 5;  // 0..3

  // Preload this wave's 16x256 Q block as 8 A-fragments (RoPE already applied).
  v16bf qf[8];
  {
    const int qrow = q0 + wv * 16 + l16;
    const ushort_t* qp = Qb + (size_t)(b * S_ + qrow) * (H_ * HD_) + (size_t)h * HD_;
#pragma unroll
    for (int kk = 0; kk < 8; ++kk) qf[kk] = load_a_frag(qp + kk * 32, hi);
  }

  v8f o[16];
#pragma unroll
  for (int t = 0; t < 16; ++t)
#pragma unroll
    for (int e = 0; e < 8; ++e) o[t][e] = 0.0f;
  float m = -3.0e38f, l = 0.0f;  // per-row state, valid in lanes 0..15

  int ks = q0 - (WIN_ - 1);
  if (ks < 0) ks = 0;
  ks &= ~31;
  const int ke = q0 + 64;

  for (int kb = ks; kb < ke; kb += 32) {
    {  // K: async copy row-major; V: synchronous transposed store
      const int krow = tid >> 2;
      const int hs = (tid & 3) << 6;
      const ushort_t* kp = Kb + (size_t)(b * S_ + kb + krow) * HD_ + hs;
      const ushort_t* vp = Vb + (size_t)(b * S_ + kb + krow) * HD_ + hs;
#pragma unroll
      for (int c = 0; c < 8; ++c)
        async_load_b128(lds_off32(&lK[krow * 264 + hs + c * 8]), kp + c * 8);
#pragma unroll
      for (int c = 0; c < 8; ++c) {
        union { uint4 q; ushort_t s[8]; } vq;
        vq.q = *(const uint4*)(vp + c * 8);
#pragma unroll
        for (int e = 0; e < 8; ++e) lVt[(hs + c * 8 + e) * 40 + krow] = vq.s[e];
      }
      if (kb + 32 < ke) {  // prefetch next V tile (K comes in via async next iter)
        __builtin_prefetch(vp + 32 * HD_, 0, 0);
      }
    }
    wait_async0();
    __syncthreads();

    // S = Q K^T (scaled, masked) -> LDS
#pragma unroll
    for (int t2 = 0; t2 < 2; ++t2) {
      v8f sc;
#pragma unroll
      for (int e = 0; e < 8; ++e) sc[e] = 0.0f;
      const int keyrow = t2 * 16 + l16;
#pragma unroll
      for (int kk = 0; kk < 8; ++kk) {
        v16bf bf = load_b_frag(&lK[keyrow * 264 + kk * 32], hi);
        sc = __builtin_amdgcn_wmma_f32_16x16x32_bf16(false, qf[kk], false, bf,
                                                     (short)0, sc, false, false);
      }
      const int key = kb + t2 * 16 + l16;
#pragma unroll
      for (int rr = 0; rr < 8; ++rr) {
        const int q = q0 + wv * 16 + rr + 8 * hi;
        const bool ok = (key <= q) && (key + (WIN_ - 1) >= q);
        lS[wv][rr + 8 * hi][t2 * 16 + l16] = ok ? sc[rr] * SCALE_ : -3.0e38f;
      }
    }
    __syncthreads();

    // Online softmax: lanes 0..15 own one query row each
    if (lane < 16) {
      const int row = lane;
      float cmax = -3.0e38f;
      float sv[32];
#pragma unroll
      for (int c = 0; c < 32; ++c) {
        sv[c] = lS[wv][row][c];
        cmax = fmaxf(cmax, sv[c]);
      }
      const float mnew = fmaxf(m, cmax);
      const float alpha = __expf(m - mnew);  // m==mnew -> 1 exactly
      float sum = 0.0f;
#pragma unroll
      for (int c = 0; c < 32; ++c) {
        const float p = (sv[c] <= -1.0e37f) ? 0.0f : __expf(sv[c] - mnew);
        sum += p;
        lP[wv][row * 40 + c] = f2bf(p);
      }
      l = l * alpha + sum;
      m = mnew;
      lAlpha[wv][row] = alpha;
    }
    __syncthreads();

    // Rescale accumulators, then O += P V
    float af8[8];
#pragma unroll
    for (int rr = 0; rr < 8; ++rr) af8[rr] = lAlpha[wv][rr + 8 * hi];
#pragma unroll
    for (int t = 0; t < 16; ++t)
#pragma unroll
      for (int rr = 0; rr < 8; ++rr) o[t][rr] *= af8[rr];

    v16bf pf = load_a_frag(&lP[wv][l16 * 40], hi);
#pragma unroll
    for (int t = 0; t < 16; ++t) {
      v16bf bf = load_b_frag(&lVt[(t * 16 + l16) * 40], hi);
      o[t] = __builtin_amdgcn_wmma_f32_16x16x32_bf16(false, pf, false, bf,
                                                     (short)0, o[t], false, false);
    }
    __syncthreads();
  }

  if (lane < 16) lL[wv][lane] = l;
  __syncthreads();
  float linv[8];
#pragma unroll
  for (int rr = 0; rr < 8; ++rr) linv[rr] = 1.0f / lL[wv][rr + 8 * hi];
#pragma unroll
  for (int t = 0; t < 16; ++t) {
#pragma unroll
    for (int rr = 0; rr < 8; ++rr) {
      const int q = q0 + wv * 16 + rr + 8 * hi;
      AO[(size_t)(b * S_ + q) * (H_ * HD_) + (size_t)h * HD_ + t * 16 + l16] =
          f2bf(o[t][rr] * linv[rr]);
    }
  }
}

static inline void launch_cvt(const float* s, ushort_t* d, int n, hipStream_t st) {
  cvt_bf16_kernel<<<(n / 4 + 255) / 256, 256, 0, st>>>(s, d, n);
}

extern "C" void kernel_launch(void* const* d_in, const int* in_sizes, int n_in,
                              void* d_out, int out_size, void* d_ws, size_t ws_size,
                              hipStream_t stream) {
  const float* hidden = (const float*)d_in[0];
  // d_in[1] attention_mask: unused — causal ∧ sliding-window is rebuilt analytically
  const int* pos = (const int*)d_in[2];
  const float* Wq = (const float*)d_in[3];
  const float* Wk = (const float*)d_in[4];
  const float* Wv = (const float*)d_in[5];
  const float* Wo = (const float*)d_in[6];

  char* w = (char*)d_ws;
  ushort_t* Xb = (ushort_t*)w;  w += (size_t)B_ * S_ * D_ * 2;        // 16 MB
  ushort_t* Wqb = (ushort_t*)w; w += (size_t)H_ * HD_ * D_ * 2;       // 8 MB
  ushort_t* Wkb = (ushort_t*)w; w += (size_t)HD_ * D_ * 2;            // 1 MB
  ushort_t* Wvb = (ushort_t*)w; w += (size_t)HD_ * D_ * 2;            // 1 MB
  ushort_t* Wob = (ushort_t*)w; w += (size_t)D_ * H_ * HD_ * 2;       // 8 MB
  ushort_t* Qb = (ushort_t*)w;  w += (size_t)B_ * S_ * H_ * HD_ * 2;  // 16 MB
  ushort_t* Kb = (ushort_t*)w;  w += (size_t)B_ * S_ * HD_ * 2;       // 2 MB
  ushort_t* Vb = (ushort_t*)w;  w += (size_t)B_ * S_ * HD_ * 2;       // 2 MB
  ushort_t* AO = (ushort_t*)w;  w += (size_t)B_ * S_ * H_ * HD_ * 2;  // 16 MB
  (void)ws_size; (void)in_sizes; (void)n_in; (void)out_size;

  const int M = B_ * S_;
  // One-shot f32 -> bf16 conversions (removes conversions + halves bytes in GEMM loops)
  launch_cvt(hidden, Xb, B_ * S_ * D_, stream);
  launch_cvt(Wq, Wqb, H_ * HD_ * D_, stream);
  launch_cvt(Wk, Wkb, HD_ * D_, stream);
  launch_cvt(Wv, Wvb, HD_ * D_, stream);
  launch_cvt(Wo, Wob, D_ * H_ * HD_, stream);

  // QKV projections (bf16 x bf16 -> bf16)
  gemm_wmma_kernel<false><<<dim3(M / 256, (H_ * HD_) / 128), 256, 0, stream>>>(
      Xb, Wqb, Qb, M, H_ * HD_, D_);
  gemm_wmma_kernel<false><<<dim3(M / 256, HD_ / 128), 256, 0, stream>>>(
      Xb, Wkb, Kb, M, HD_, D_);
  gemm_wmma_kernel<false><<<dim3(M / 256, HD_ / 128), 256, 0, stream>>>(
      Xb, Wvb, Vb, M, HD_, D_);
  // RoPE (in-place)
  {
    const int totq = M * H_ * (HD_ / 2);
    rope_kernel<<<(totq + 255) / 256, 256, 0, stream>>>(Qb, pos, H_, totq);
    const int totk = M * (HD_ / 2);
    rope_kernel<<<(totk + 255) / 256, 256, 0, stream>>>(Kb, pos, 1, totk);
  }
  // Attention
  attn_kernel<<<B_ * H_ * (S_ / 64), 128, 0, stream>>>(Qb, Kb, Vb, AO);
  // Output projection (bf16 in, f32 out)
  gemm_wmma_kernel<true><<<dim3(M / 256, D_ / 128), 256, 0, stream>>>(
      AO, Wob, (float*)d_out, M, D_, H_ * HD_);
}
